// GCNNet_54382875902525
// MI455X (gfx1250) — matrix operand order repaired and checked by reading, not compiled, not run on hardware
//
#include <hip/hip_runtime.h>
#include <hip/hip_bf16.h>

typedef __attribute__((ext_vector_type(16))) _Float16 v16h;
typedef __attribute__((ext_vector_type(8)))  _Float16 v8h;
typedef __attribute__((ext_vector_type(8)))  float    v8f;
typedef __attribute__((ext_vector_type(4)))  float    v4f;

#define TPB 256

__device__ __forceinline__ long gidx() {
  return (long)blockIdx.x * blockDim.x + threadIdx.x;
}

// ---------- elementwise / prep kernels ----------

__global__ void copy_pad_f32(const float* __restrict__ src, float* __restrict__ dst,
                             int rows, int cols, int ld) {
  long idx = gidx();
  long total = (long)rows * ld;
  if (idx >= total) return;
  int  f = (int)(idx % ld);
  long v = idx / ld;
  dst[idx] = (f < cols) ? src[v * (long)cols + f] : 0.0f;
}

__global__ void deg_init(float* __restrict__ deg, int n) {
  long i = gidx();
  if (i < n) deg[i] = 1.0f;  // self loop
}

__global__ void deg_edges(const int* __restrict__ dst, float* __restrict__ deg, int E) {
  long e = gidx();
  if (e < E) atomicAdd(&deg[dst[e]], 1.0f);
}

__global__ void deg_finalize(float* __restrict__ deg, int n) {
  long i = gidx();
  if (i < n) deg[i] = rsqrtf(deg[i]);  // deg >= 1 always
}

// Agg[v, f4..f4+3] = dinv[v]^2 * X[v, f4..f4+3]   (ld % 4 == 0; pads stay 0)
__global__ void agg_init(const float* __restrict__ X, const float* __restrict__ dinv,
                         float* __restrict__ Agg, int n, int ld4) {
  long idx = gidx();
  long total = (long)n * ld4;
  if (idx >= total) return;
  long v = idx / ld4;
  float di = dinv[v];
  v4f x = *(const v4f*)(X + idx * 4);
  v4f r = {di * di * x.x, di * di * x.y, di * di * x.z, di * di * x.w};
  *(v4f*)(Agg + idx * 4) = r;
}

// Agg[d, f4..f4+3] += dinv[s]*dinv[d] * X[s, f4..f4+3]; K rounded up to mult of 4
// (X pad columns are 0, so over-reach into pads adds 0.0 into zeroed Agg pads).
__global__ void edge_scatter(const int* __restrict__ src, const int* __restrict__ dst,
                             const float* __restrict__ dinv,
                             const float* __restrict__ X, float* __restrict__ Agg,
                             long E, int KB, int ld) {
  long idx = gidx();
  long total = E * KB;
  if (idx >= total) return;
  long e  = idx / KB;
  int  f4 = (int)(idx % KB) * 4;
  int  s = src[e], d = dst[e];
  float c = dinv[s] * dinv[d];
  v4f x = *(const v4f*)(X + (long)s * ld + f4);
  float* a = Agg + (long)d * ld + f4;
  atomicAdd(a + 0, c * x.x);
  atomicAdd(a + 1, c * x.y);
  atomicAdd(a + 2, c * x.z);
  atomicAdd(a + 3, c * x.w);
}

__global__ void f32_to_f16(const float* __restrict__ src, _Float16* __restrict__ dst, long total) {
  long idx = gidx();
  if (idx < total) dst[idx] = (_Float16)src[idx];
}

// X[v,f] = relu(Hw[v,f] + b[f]) for f < N, else 0 (keeps pad columns zero); float4
__global__ void bias_relu(const float* __restrict__ Hw, const float* __restrict__ b,
                          float* __restrict__ X, int n, int ld4, int N) {
  long idx = gidx();
  long total = (long)n * ld4;
  if (idx >= total) return;
  int f4 = (int)(idx % ld4) * 4;
  v4f h = *(const v4f*)(Hw + idx * 4);
  v4f r;
  r.x = (f4 + 0 < N) ? fmaxf(h.x + b[f4 + 0], 0.0f) : 0.0f;
  r.y = (f4 + 1 < N) ? fmaxf(h.y + b[f4 + 1], 0.0f) : 0.0f;
  r.z = (f4 + 2 < N) ? fmaxf(h.z + b[f4 + 2], 0.0f) : 0.0f;
  r.w = (f4 + 3 < N) ? fmaxf(h.w + b[f4 + 3], 0.0f) : 0.0f;
  *(v4f*)(X + idx * 4) = r;
}

// Pack W[K,N] f32 row-major into WMMA B-fragment order (f16, zero padded):
// Wp[((kb*tilesN + nt)*32 + lane)*16 + p] = W[kb*32 + (lane<16?0:16) + p][nt*16 + lane%16]
__global__ void pack_w(const float* __restrict__ W, _Float16* __restrict__ Wp,
                       int K, int N, int kblocks, int tilesN) {
  long idx = gidx();
  long total = (long)kblocks * tilesN * 512;
  if (idx >= total) return;
  int  p    = (int)(idx & 15);
  int  lane = (int)((idx >> 4) & 31);
  long t    = idx >> 9;
  int  nt   = (int)(t % tilesN);
  int  kb   = (int)(t / tilesN);
  int  n    = nt * 16 + (lane & 15);
  int  kr   = kb * 32 + ((lane < 16) ? 0 : 16) + p;
  float v = (kr < K && n < N) ? W[(long)kr * N + n] : 0.0f;
  Wp[idx] = (_Float16)v;
}

__global__ void pool_init(int* __restrict__ pooled, long total) {
  long idx = gidx();
  if (idx < total) pooled[idx] = 0;  // +0.0f bits, valid identity post-ReLU
}

// global max pool via int-reinterpret atomicMax (all values >= 0)
__global__ void pool_max(const float* __restrict__ X, const int* __restrict__ batch,
                         int* __restrict__ pooled, int n, int ld) {
  long idx = gidx();
  long total = (long)n * ld;
  if (idx >= total) return;
  long v = idx / ld;
  int  f = (int)(idx % ld);
  atomicMax(&pooled[(long)batch[v] * ld + f], __float_as_int(X[idx]));
}

__global__ void pooled_to_f16(const int* __restrict__ pooled, _Float16* __restrict__ dst, long total) {
  long idx = gidx();
  if (idx < total) dst[idx] = (_Float16)__int_as_float(pooled[idx]);
}

// ---------- WMMA GEMM: C[M,ldc] f32 = A[M,lda] f16 @ Bpacked (+bias, relu) ----------
// Each wave computes one 16x16 M-tile x TWO adjacent N-tiles (tilesN is always even
// here), so the A fragment is loaded once per K-block and feeds two WMMAs.
// M%16==0, K=kblocks*32 zero-padded, N = tilesN*16 == ldc.
// Early exit is wave-uniform -> EXEC stays all-1s around WMMA.
__global__ __launch_bounds__(TPB)
void wmma_gemm(const _Float16* __restrict__ A, const _Float16* __restrict__ Bp,
               float* __restrict__ C, const float* __restrict__ bias,
               int M, int lda, int ldc, int kblocks, int tilesN, int relu) {
  int lane = threadIdx.x & 31;
  int wave = threadIdx.x >> 5;
  int tilesN2 = tilesN >> 1;
  long tile = (long)blockIdx.x * 8 + wave;
  long ntiles = (long)(M >> 4) * tilesN2;
  if (tile >= ntiles) return;
  int tm  = (int)(tile / tilesN2);
  int tn0 = (int)(tile % tilesN2) * 2;

  int m = tm * 16 + (lane & 15);
  int koff = (lane < 16) ? 0 : 16;                   // contiguous K per lane half (matches pack_w)
  const _Float16* arow  = A + (long)m * lda;
  const _Float16* bb0 = Bp + ((long)tn0 * 32 + lane) * 16;
  const _Float16* bb1 = bb0 + 32 * 16;               // next N-tile
  long bstride = (long)tilesN * 512;

  v8f acc0 = {}, acc1 = {};
  for (int kb = 0; kb < kblocks; ++kb) {
    int k0 = kb * 32;
    union { v16h v; v8h h[2]; } a;
    a.h[0] = *(const v8h*)(arow + k0 + koff);        // K = k0+koff .. +7
    a.h[1] = *(const v8h*)(arow + k0 + koff + 8);    // K = k0+koff+8 .. +15
    v16h b0 = *(const v16h*)(bb0 + kb * bstride);
    v16h b1 = *(const v16h*)(bb1 + kb * bstride);
    acc0 = __builtin_amdgcn_wmma_f32_16x16x32_f16(false, a.v, false, b0,
                                                  (short)0, acc0, false, false);
    acc1 = __builtin_amdgcn_wmma_f32_16x16x32_f16(false, a.v, false, b1,
                                                  (short)0, acc1, false, false);
  }

  int n0 = tn0 * 16 + (lane & 15);
  float bv0 = bias ? bias[n0] : 0.0f;
  float bv1 = bias ? bias[n0 + 16] : 0.0f;
  int rbase = tm * 16 + ((lane < 16) ? 0 : 8);       // C fragment rows: i / i+8
#pragma unroll
  for (int i = 0; i < 8; ++i) {
    float v0 = acc0[i] + bv0;
    float v1 = acc1[i] + bv1;
    if (relu) { v0 = fmaxf(v0, 0.0f); v1 = fmaxf(v1, 0.0f); }
    float* crow = C + (long)(rbase + i) * ldc;
    crow[n0] = v0;
    crow[n0 + 16] = v1;
  }
}

// ---------- launcher ----------

static inline int nblk(long total) { return (int)((total + TPB - 1) / TPB); }
static inline int gemm_blocks(int M, int tilesN) {
  long ntiles = (long)(M >> 4) * (tilesN >> 1);
  return (int)((ntiles + 7) / 8);
}

extern "C" void kernel_launch(void* const* d_in, const int* in_sizes, int n_in,
                              void* d_out, int out_size, void* d_ws, size_t ws_size,
                              hipStream_t stream) {
  (void)in_sizes; (void)n_in; (void)out_size; (void)ws_size;
  const int NN = 50000, NE = 800000, NG = 256;
  const int P0 = 96, P1 = 96, P2 = 160, P3 = 320;   // padded widths 75/75/150/300

  const float* x   = (const float*)d_in[0];
  const int*   ei  = (const int*)d_in[1];
  const int*   src = ei;
  const int*   dst = ei + NE;
  const int*   bat = (const int*)d_in[2];
  const float* W1  = (const float*)d_in[3],  *b1  = (const float*)d_in[4];
  const float* W2  = (const float*)d_in[5],  *b2  = (const float*)d_in[6];
  const float* W3  = (const float*)d_in[7],  *b3  = (const float*)d_in[8];
  const float* Wg1 = (const float*)d_in[9],  *bg1 = (const float*)d_in[10];
  const float* Wg2 = (const float*)d_in[11], *bg2 = (const float*)d_in[12];
  float* out = (float*)d_out;

  // workspace carve-up (every chunk size is a multiple of 32 B)
  char* ws = (char*)d_ws;
  size_t off = 0;
  float*    X     = (float*)(ws + off);    off += (size_t)NN * P3 * 4;       // features f32
  float*    AggHw = (float*)(ws + off);    off += (size_t)NN * P3 * 4;       // Agg, then GEMM out
  _Float16* Aggh  = (_Float16*)(ws + off); off += (size_t)NN * P2 * 2;       // f16 GEMM input
  float*    dinv  = (float*)(ws + off);    off += (size_t)NN * 4;
  _Float16* Wp    = (_Float16*)(ws + off); off += (size_t)10 * 64 * 512 * 2; // max packed W (fc_g1)
  int*      pool  = (int*)(ws + off);      off += (size_t)NG * P3 * 4;
  _Float16* Ph    = (_Float16*)(ws + off); off += (size_t)NG * P3 * 2;
  float*    G1    = (float*)(ws + off);    off += (size_t)NG * 1024 * 4;
  _Float16* G1h   = (_Float16*)(ws + off); off += (size_t)NG * 1024 * 2;

  // 1) pad input features
  copy_pad_f32<<<nblk((long)NN * P0), TPB, 0, stream>>>(x, X, NN, 75, P0);

  // 2) dinv = rsqrt(1 + indeg)
  deg_init<<<nblk(NN), TPB, 0, stream>>>(dinv, NN);
  deg_edges<<<nblk(NE), TPB, 0, stream>>>(dst, dinv, NE);
  deg_finalize<<<nblk(NN), TPB, 0, stream>>>(dinv, NN);

  // 3) GCN layers, aggregate-then-transform: relu( (A_hat X) W + b )
  const float* LW[3] = {W1, W2, W3};
  const float* Lb[3] = {b1, b2, b3};
  const int Kin[3]   = {75, 75, 150};
  const int ldin[3]  = {P0, P1, P2};
  const int Nout[3]  = {75, 150, 300};
  const int ldout[3] = {P1, P2, P3};
  for (int l = 0; l < 3; ++l) {
    int kblocks = ldin[l] / 32, tilesN = ldout[l] / 16;
    int KB = (Kin[l] + 3) / 4;  // float4 blocks per edge (over-reach lands in zero pads)
    agg_init<<<nblk((long)NN * (ldin[l] / 4)), TPB, 0, stream>>>(
        X, dinv, AggHw, NN, ldin[l] / 4);
    edge_scatter<<<nblk((long)NE * KB), TPB, 0, stream>>>(
        src, dst, dinv, X, AggHw, NE, KB, ldin[l]);
    f32_to_f16<<<nblk((long)NN * ldin[l]), TPB, 0, stream>>>(AggHw, Aggh, (long)NN * ldin[l]);
    pack_w<<<nblk((long)kblocks * tilesN * 512), TPB, 0, stream>>>(
        LW[l], Wp, Kin[l], Nout[l], kblocks, tilesN);
    wmma_gemm<<<gemm_blocks(NN, tilesN), TPB, 0, stream>>>(
        Aggh, Wp, AggHw, nullptr, NN, ldin[l], ldout[l], kblocks, tilesN, 0);
    bias_relu<<<nblk((long)NN * (ldout[l] / 4)), TPB, 0, stream>>>(
        AggHw, Lb[l], X, NN, ldout[l] / 4, Nout[l]);
  }

  // 4) global max pool over graphs -> [256, 320] (pads stay 0)
  pool_init<<<nblk((long)NG * P3), TPB, 0, stream>>>(pool, (long)NG * P3);
  pool_max<<<nblk((long)NN * P3), TPB, 0, stream>>>(X, bat, pool, NN, P3);
  pooled_to_f16<<<nblk((long)NG * P3), TPB, 0, stream>>>(pool, Ph, (long)NG * P3);

  // 5) fc_g1: [256,320] @ [320,1024] + bias, relu
  pack_w<<<nblk((long)10 * 64 * 512), TPB, 0, stream>>>(Wg1, Wp, 300, 1024, 10, 64);
  wmma_gemm<<<gemm_blocks(NG, 64), TPB, 0, stream>>>(Ph, Wp, G1, bg1, NG, P3, 1024, 10, 64, 1);
  f32_to_f16<<<nblk((long)NG * 1024), TPB, 0, stream>>>(G1, G1h, (long)NG * 1024);

  // 6) fc_g2: [256,1024] @ [1024,128] + bias -> d_out
  pack_w<<<nblk((long)32 * 8 * 512), TPB, 0, stream>>>(Wg2, Wp, 1024, 128, 32, 8);
  wmma_gemm<<<gemm_blocks(NG, 8), TPB, 0, stream>>>(G1h, Wp, out, bg2, NG, 1024, 128, 32, 8, 0);
}
